// S2CNNCls_21079699489119
// MI455X (gfx1250) — matrix operand order repaired and checked by reading, not compiled, not run on hardware
//
#include <hip/hip_runtime.h>
#include <hip/hip_bf16.h>
#include <math.h>

// ---------------------------------------------------------------------------
// S2CNN forward for MI455X (gfx1250, wave32).
// Hot matrix op (SO3 correlation, feature-contracted complex GEMM) runs on
// v_wmma_f32_16x16x32_f16 with operands pre-packed in fragment-ready f16
// layout (one 32B v16h load per lane per operand per K-step). Everything else
// is bandwidth-bound f32 with LDS-tiled DFTs.
// Required ws_size: ~100M floats (~400 MB).
// ---------------------------------------------------------------------------

#define PI_F 3.14159265358979323846f
#define PI_D 3.14159265358979323846

typedef __attribute__((ext_vector_type(16))) _Float16 v16h;
typedef __attribute__((ext_vector_type(8)))  float    v8f;

static __device__ __forceinline__ float2 cmulf2(float2 a, float2 b){ return make_float2(a.x*b.x - a.y*b.y, a.x*b.y + a.y*b.x); }
static __device__ __forceinline__ float2 caddf2(float2 a, float2 b){ return make_float2(a.x+b.x, a.y+b.y); }
static __device__ __forceinline__ float2 csclf2(float2 a, float s){ return make_float2(a.x*s, a.y*s); }
static __device__ __forceinline__ float2 cisf(float t){ float s,c; __sincosf(t,&s,&c); return make_float2(c,s); }

// sum_{j<l} (2j+1)^2  (packed offset of block l)
static __host__ __device__ __forceinline__ int offsq(int l){ return (4*l*l*l - l)/3; }

// padded GEMM dims per l (R=8w rows, C=K=100w)
static __host__ __device__ __forceinline__ int rp16(int w){ int R = 8*w;  return ((R + 15)/16)*16; }
static __host__ __device__ __forceinline__ int cp16(int w){ int C = 100*w; return ((C + 15)/16)*16; }
static __host__ __device__ __forceinline__ int kp32(int w){ int K = 100*w; return ((K + 31)/32)*32; }
static __host__ __device__ size_t apack_off(int l){ size_t o = 0; for (int j = 0; j < l; ++j){ int w = 2*j+1; o += (size_t)rp16(w)*kp32(w); } return o; }
static __host__ __device__ size_t bpack_off(int l){ size_t o = 0; for (int j = 0; j < l; ++j){ int w = 2*j+1; o += (size_t)cp16(w)*kp32(w); } return o; }

static __device__ __forceinline__ void dec_lmn(int idx, int* pl, int* pm, int* pn){
  int l = 0, off = 0;
  for (;;){ int sz = (2*l+1)*(2*l+1); if (idx < off + sz) break; off += sz; ++l; }
  int r = idx - off, w = 2*l+1;
  *pl = l; *pm = r / w; *pn = r % w;
}

// Wigner-d d^l_{mp,mm}(beta), closed-form Jacobi sum (double; l<=15 -> 30! fits)
static __device__ double wigd(int l, int mp, int mm, double beta){
  double fact[64]; fact[0] = 1.0;
  for (int i = 1; i <= 2*l; ++i) fact[i] = fact[i-1]*(double)i;
  double cb = cos(0.5*beta), sb = sin(0.5*beta);
  int smin = mm - mp; if (smin < 0) smin = 0;
  int smax = l + mm; if (l - mp < smax) smax = l - mp;
  double pref = sqrt(fact[l+mp]*fact[l-mp]*fact[l+mm]*fact[l-mm]);
  double sum = 0.0;
  for (int s = smin; s <= smax; ++s){
    double den = fact[l+mm-s]*fact[s]*fact[mp-mm+s]*fact[l-mp-s];
    double sgn = ((mp-mm+s)&1) ? -1.0 : 1.0;
    sum += sgn/den * pow(cb, (double)(2*l+mm-mp-2*s)) * pow(sb, (double)(mp-mm+2*s));
  }
  return pref*sum;
}

// ---------------------------------------------------------------------------
// Table builders
// ---------------------------------------------------------------------------
__global__ void k_qw(float* qw, int b){
  int k = blockIdx.x*blockDim.x + threadIdx.x;
  if (k >= 2*b) return;
  double beta = PI_D*(2*k+1)/(4.0*b);
  double s = 0.0;
  for (int j = 0; j < b; ++j) s += sin((double)(2*k+1)*(2*j+1)*PI_D/(4.0*b)) / (double)(2*j+1);
  qw[k] = (float)((2.0/b) * sin(beta) * s);
}

// d^l_{m,0}(beta_k) on DH(30) grid; packed [k][l*l + (m+l)], k<60, l<16
__global__ void k_wig_d0(float* out){
  int t = blockIdx.x*blockDim.x + threadIdx.x;
  if (t >= 60*256) return;
  int k = t / 256, idx = t % 256;
  int l = 0; while ((l+1)*(l+1) <= idx) ++l;
  int m = idx - l*l - l;
  double beta = PI_D*(2*k+1)/120.0;
  out[t] = (float)wigd(l, m, 0, beta);
}

// full d^l(beta_k), packed [k][offsq(l) + m0*(2l+1)+n0]; half=1 -> beta=pi/2
__global__ void k_wig_pack(float* out, int nbeta, int pack, int bgrid, int half){
  int t = blockIdx.x*blockDim.x + threadIdx.x;
  if (t >= nbeta*pack) return;
  int k = t / pack, idx = t % pack;
  int l, m0, n0; dec_lmn(idx, &l, &m0, &n0);
  double beta = half ? (PI_D*0.5) : (PI_D*(2*k+1)/(4.0*bgrid));
  out[t] = (float)wigd(l, m0 - l, n0 - l, beta);
}

// ---------------------------------------------------------------------------
// Stage 1: project point cloud to S2 grid (scatter-max)
// ---------------------------------------------------------------------------
__global__ void k_project(const float* __restrict__ x, float* __restrict__ img){
  int t = blockIdx.x*blockDim.x + threadIdx.x;
  if (t >= 8*1024) return;
  int b = t >> 10, n = t & 1023;
  float px = x[(b*3+0)*1024+n], py = x[(b*3+1)*1024+n], pz = x[(b*3+2)*1024+n];
  float r = sqrtf(px*px + py*py + pz*pz);
  float ct = pz / fmaxf(r, 1e-8f);
  ct = fminf(fmaxf(ct, -1.f), 1.f);
  float beta = acosf(ct);
  float alpha = atan2f(py, px); if (alpha < 0.f) alpha += 2.f*PI_F;
  int bi = (int)(beta / PI_F * 60.f);        bi = min(max(bi, 0), 59);
  int ai = (int)(alpha / (2.f*PI_F) * 60.f); ai = min(max(ai, 0), 59);
  atomicMax((unsigned int*)&img[(b*60+bi)*60+ai], __float_as_uint(r));  // r >= 0
}

// DFT over alpha: G[b,k,f] = sum_a img[b,k,a] e^{-2pi i f a/60}
__global__ void k_dft_alpha(const float* __restrict__ img, float2* __restrict__ G){
  int t = blockIdx.x*blockDim.x + threadIdx.x;
  if (t >= 8*60*60) return;
  int f = t % 60, bk = t / 60;
  const float* row = img + bk*60;
  float2 acc = make_float2(0.f, 0.f);
  for (int a = 0; a < 60; ++a) acc = caddf2(acc, csclf2(cisf(-2.f*PI_F*(float)f*(float)a/60.f), row[a]));
  G[t] = acc;
}

// X[b, l^2+m0] = (2pi/60) sum_k qw30[k] d0[k, l^2+m0] G[b,k, m mod 60]
__global__ void k_s2fft(const float2* __restrict__ G, const float* __restrict__ qw30,
                        const float* __restrict__ d0, float2* __restrict__ X){
  int t = blockIdx.x*blockDim.x + threadIdx.x;
  if (t >= 8*256) return;
  int b = t >> 8, idx = t & 255;
  int l = 0; while ((l+1)*(l+1) <= idx) ++l;
  int m = idx - l*l - l;
  int mc = ((m % 60) + 60) % 60;
  float2 acc = make_float2(0.f, 0.f);
  for (int k = 0; k < 60; ++k)
    acc = caddf2(acc, csclf2(G[(b*60+k)*60 + mc], qw30[k]*d0[k*256+idx]));
  X[t] = csclf2(acc, 2.f*PI_F/60.f);
}

// Y[o, l^2+m0] = d^l_{m,0}(pi/2) * (1/60) sum_p w_s2[o,p] e^{-i m alpha_p}
__global__ void k_s2rft(const float* __restrict__ w_s2, const float* __restrict__ dhalf,
                        float2* __restrict__ Y){
  int t = blockIdx.x*blockDim.x + threadIdx.x;
  if (t >= 100*256) return;
  int o = t >> 8, idx = t & 255;
  int l = 0; while ((l+1)*(l+1) <= idx) ++l;
  int m = idx - l*l - l;
  float dv = dhalf[offsq(l) + (m+l)*(2*l+1) + l];   // column n=0
  float2 acc = make_float2(0.f, 0.f);
  for (int p = 0; p < 60; ++p)
    acc = caddf2(acc, csclf2(cisf(-(float)m * 2.f*PI_F*(float)p/60.f), w_s2[o*60+p]));
  Y[t] = csclf2(acc, dv/60.f);
}

// Z[b,o, offsq(l)+m0*w+n0] = X[b,l,m] * conj(Y[o,l,n])
__global__ void k_zouter(const float2* __restrict__ X, const float2* __restrict__ Y,
                         float2* __restrict__ Z){
  int t = blockIdx.x*blockDim.x + threadIdx.x;
  if (t >= 800*5456) return;
  int pair = t / 5456, idx = t % 5456;
  int b = pair / 100, o = pair % 100;
  int l, m0, n0; dec_lmn(idx, &l, &m0, &n0);
  float2 xv = X[b*256 + l*l + m0];
  float2 yv = Y[o*256 + l*l + n0];
  yv.y = -yv.y;
  Z[t] = cmulf2(xv, yv);
}

// Spectral scatter for so3_ifft: M[pair,k,mm,nn] = sum_l (2l+1) d^l_k[m,n] Z[pair,l,m,n]
__global__ void k_maccum(const float2* __restrict__ Z, const float* __restrict__ dtab,
                         float2* __restrict__ M, int N, int Lmax, int pack, long long total){
  long long t = (long long)blockIdx.x*blockDim.x + threadIdx.x;
  if (t >= total) return;
  int NN = N*N;
  int cell = (int)(t % NN);
  long long slab = t / NN;
  int k = (int)(slab % N);
  long long pair = slab / N;
  int mm = cell / N, nn = cell % N;
  float2 acc = make_float2(0.f, 0.f);
  for (int l = 0; l < Lmax; ++l){
    int w = 2*l+1;
    int mc = (mm + l) % N; if (mc >= w) continue;
    int nc = (nn + l) % N; if (nc >= w) continue;
    int o = offsq(l) + mc*w + nc;
    float dv = (float)(2*l+1) * dtab[k*pack + o];
    acc = caddf2(acc, csclf2(Z[pair*pack + o], dv));
  }
  M[slab*NN + cell] = acc;
}

// Unnormalized inverse DFT2 of one NxN spectral tile -> real + per-channel bias
template<int N>
__global__ void k_itile(const float2* __restrict__ M, const float* __restrict__ bias,
                        float* __restrict__ out){
  __shared__ float2 s0[N*N];
  __shared__ float2 s1[N*N];
  const int tid = threadIdx.x, bd = blockDim.x;
  const size_t base = (size_t)blockIdx.x * (N*N);
  for (int i = tid; i < N*N; i += bd) s0[i] = M[base + i];
  __syncthreads();
  for (int i = tid; i < N*N; i += bd){
    int m = i / N, g = i % N;
    float2 acc = make_float2(0.f, 0.f);
    for (int n = 0; n < N; ++n)
      acc = caddf2(acc, cmulf2(s0[m*N+n], cisf(2.f*PI_F*(float)g*(float)n/(float)N)));
    s1[i] = acc;
  }
  __syncthreads();
  const int f = (blockIdx.x / N) % 100;
  const float bf = bias[f];
  for (int i = tid; i < N*N; i += bd){
    int a = i / N, g = i % N;
    float acc = 0.f;
    for (int m = 0; m < N; ++m){
      float2 e = cisf(2.f*PI_F*(float)a*(float)m/(float)N);
      float2 v = s1[m*N+g];
      acc += v.x*e.x - v.y*e.y;
    }
    out[base + i] = acc + bf;
  }
}

// Forward DFT2 of one NxN real tile -> complex spectrum
template<int N>
__global__ void k_ftile(const float* __restrict__ in, float2* __restrict__ out){
  __shared__ float  s0[N*N];
  __shared__ float2 s1[N*N];
  const int tid = threadIdx.x, bd = blockDim.x;
  const size_t base = (size_t)blockIdx.x * (N*N);
  for (int i = tid; i < N*N; i += bd) s0[i] = in[base + i];
  __syncthreads();
  for (int i = tid; i < N*N; i += bd){
    int a = i / N, nf = i % N;
    float2 acc = make_float2(0.f, 0.f);
    for (int g = 0; g < N; ++g)
      acc = caddf2(acc, csclf2(cisf(-2.f*PI_F*(float)nf*(float)g/(float)N), s0[a*N+g]));
    s1[i] = acc;
  }
  __syncthreads();
  for (int i = tid; i < N*N; i += bd){
    int mf = i / N, nf = i % N;
    float2 acc = make_float2(0.f, 0.f);
    for (int a = 0; a < N; ++a)
      acc = caddf2(acc, cmulf2(s1[a*N+nf], cisf(-2.f*PI_F*(float)mf*(float)a/(float)N)));
    out[base + i] = acc;
  }
}

// Per-channel mean/var over batch+spatial
__global__ void k_bnstats(const float* __restrict__ x, float* __restrict__ mu,
                          float* __restrict__ var, int S, int Bn){
  const int f = blockIdx.x, tid = threadIdx.x;
  __shared__ float s1[256], s2[256];
  float a1 = 0.f, a2 = 0.f;
  for (int b = 0; b < Bn; ++b){
    const float* p = x + ((size_t)(b*100+f))*S;
    for (int i = tid; i < S; i += 256){ float v = p[i]; a1 += v; a2 += v*v; }
  }
  s1[tid] = a1; s2[tid] = a2; __syncthreads();
  for (int st = 128; st > 0; st >>= 1){
    if (tid < st){ s1[tid] += s1[tid+st]; s2[tid] += s2[tid+st]; }
    __syncthreads();
  }
  if (tid == 0){
    float inv = 1.f / ((float)Bn * (float)S);
    float m = s1[0]*inv;
    mu[f] = m; var[f] = s2[0]*inv - m*m;
  }
}

__global__ void k_bnrelu(float* __restrict__ x, const float* __restrict__ mu,
                         const float* __restrict__ var, const float* __restrict__ g,
                         const float* __restrict__ b, int S, long long total){
  long long t = (long long)blockIdx.x*blockDim.x + threadIdx.x;
  if (t >= total) return;
  int f = (int)((t / S) % 100);
  float y = (x[t] - mu[f]) * rsqrtf(var[f] + 1e-5f) * g[f] + b[f];
  x[t] = fmaxf(y, 0.f);
}

// so3_fft contraction over beta -> fragment-packed f16 A operand
// A[r=(b,m)][t=(i,k)]; contiguous-k lane layout (k = 16*(lane/16) + elem), 16x32 tiles.
__global__ void k_x2a(const float2* __restrict__ G2, const float* __restrict__ qw16,
                      const float* __restrict__ d32,
                      _Float16* __restrict__ Apr, _Float16* __restrict__ Api){
  int t = blockIdx.x*blockDim.x + threadIdx.x;
  if (t >= 800*1330) return;
  int pair = t / 1330, idx = t % 1330;
  int l, m0, n0; dec_lmn(idx, &l, &m0, &n0);
  int w = 2*l+1;
  int mi = ((m0 - l) % 32 + 32) % 32;
  int ni = ((n0 - l) % 32 + 32) % 32;
  float2 acc = make_float2(0.f, 0.f);
  for (int k = 0; k < 32; ++k)
    acc = caddf2(acc, csclf2(G2[((size_t)pair*32 + k)*1024 + mi*32 + ni],
                             qw16[k]*d32[k*5456 + offsq(l) + m0*w + n0]));
  float sc = 2.f*PI_F/32.f; sc *= sc;
  acc = csclf2(acc, sc);
  int b = pair / 100, c = pair % 100;
  int r  = b*w + m0;               // GEMM row
  int kk = c*w + n0;               // GEMM reduction index
  int tr = r >> 4, lane15 = r & 15;
  int ks = kk >> 5, ko = kk & 31;
  int lane = lane15 + ((ko & 16) ? 16 : 0);
  int j = ko & 15;
  size_t pos = apack_off(l) + (((size_t)tr*(kp32(w) >> 5) + ks)*32 + lane)*16 + j;
  Apr[pos] = (_Float16)acc.x; Api[pos] = (_Float16)acc.y;
}

// Wf[i,o,dm] = (1/32) sum_p w_so3[i,o,p] e^{-i dm alpha_p}, dm in [-9,9]
__global__ void k_wf(const float* __restrict__ w_so3, float2* __restrict__ Wf){
  int t = blockIdx.x*blockDim.x + threadIdx.x;
  if (t >= 100*100*19) return;
  int io = t / 19, dm = t % 19 - 9;
  float2 acc = make_float2(0.f, 0.f);
  for (int p = 0; p < 32; ++p)
    acc = caddf2(acc, csclf2(cisf(-(float)dm * 2.f*PI_F*(float)p/32.f), w_so3[io*32+p]));
  Wf[t] = csclf2(acc, 1.f/32.f);
}

// Y2 spectral kernel -> fragment-packed f16 B operand
// B[t=(i,k=nu)][c=(o,n=mu)]; contiguous-k lane layout (k = 16*(lane/16) + elem), 32x16 tiles.
__global__ void k_y2b(const float* __restrict__ dhalf, const float2* __restrict__ Wf,
                      _Float16* __restrict__ Bpr, _Float16* __restrict__ Bpi){
  int t = blockIdx.x*blockDim.x + threadIdx.x;
  if (t >= 10000*1330) return;
  int io = t / 1330, idx = t % 1330;
  int l, mu0, nu0; dec_lmn(idx, &l, &mu0, &nu0);
  int w = 2*l+1;
  float dv = dhalf[offsq(l) + mu0*w + nu0];
  float2 val = csclf2(Wf[io*19 + (mu0 - l) + 9], dv);
  int i = io / 100, o = io % 100;
  int cc = o*w + mu0;              // GEMM column
  int kk = i*w + nu0;              // GEMM reduction index
  int tc = cc >> 4, lane15 = cc & 15;
  int ks = kk >> 5, ko = kk & 31;
  int lane = lane15 + ((ko & 16) ? 16 : 0);
  int j = ko & 15;
  size_t pos = bpack_off(l) + (((size_t)tc*(kp32(w) >> 5) + ks)*32 + lane)*16 + j;
  Bpr[pos] = (_Float16)val.x; Bpi[pos] = (_Float16)val.y;
}

// ---------------------------------------------------------------------------
// Complex GEMM via WMMA on fragment-packed operands.
//   Zr = Ar.Br + Ai.Bi ; Zi = Ai.Br - Ar.Bi  (4x v_wmma_f32_16x16x32_f16)
// One wave32 per 16x16 output tile; one v16h (32B) load per operand per step.
// ---------------------------------------------------------------------------
__global__ __launch_bounds__(32)
void k_zgemm(const _Float16* __restrict__ Apr, const _Float16* __restrict__ Api,
             const _Float16* __restrict__ Bpr, const _Float16* __restrict__ Bpi,
             float2* __restrict__ Z, int l, int R, int C, int KS){
  const int lane = threadIdx.x & 31;
  const int tr = blockIdx.y, tc = blockIdx.x;
  const v16h* __restrict__ par = (const v16h*)Apr + (size_t)tr*KS*32 + lane;
  const v16h* __restrict__ pai = (const v16h*)Api + (size_t)tr*KS*32 + lane;
  const v16h* __restrict__ pbr = (const v16h*)Bpr + (size_t)tc*KS*32 + lane;
  const v16h* __restrict__ pbi = (const v16h*)Bpi + (size_t)tc*KS*32 + lane;
  v8f accR = {}, accI = {};
  for (int ks = 0; ks < KS; ++ks){
    const size_t o = (size_t)ks*32;
    v16h ar = par[o];
    v16h ai = pai[o];
    v16h br = pbr[o];
    v16h bi = pbi[o];
    __builtin_prefetch((const void*)(par + o + 32), 0, 1);
    __builtin_prefetch((const void*)(pbr + o + 32), 0, 1);
    v16h bin;
#pragma unroll
    for (int j = 0; j < 16; ++j) bin[j] = -bi[j];
    accR = __builtin_amdgcn_wmma_f32_16x16x32_f16(false, ar, false, br,  (short)0, accR, false, false);
    accR = __builtin_amdgcn_wmma_f32_16x16x32_f16(false, ai, false, bi,  (short)0, accR, false, false);
    accI = __builtin_amdgcn_wmma_f32_16x16x32_f16(false, ai, false, br,  (short)0, accI, false, false);
    accI = __builtin_amdgcn_wmma_f32_16x16x32_f16(false, ar, false, bin, (short)0, accI, false, false);
  }
  const int w = 2*l+1;
#pragma unroll
  for (int v = 0; v < 8; ++v){
    int r = tr*16 + v + ((lane & 16) ? 8 : 0);
    int c = tc*16 + (lane & 15);
    if (r < R && c < C){
      int bb = r / w, m = r % w;
      int oo = c / w, n = c % w;
      Z[(size_t)(bb*100 + oo)*1330 + offsq(l) + m*w + n] = make_float2(accR[v], accI[v]);
    }
  }
}

// Integrate over SO3 grid with beta quadrature
__global__ void k_integrate(const float* __restrict__ h2, const float* __restrict__ qw10,
                            float* __restrict__ feat){
  const int bf = blockIdx.x, tid = threadIdx.x;
  __shared__ float s[256];
  const float* p = h2 + (size_t)bf*8000;
  float a = 0.f;
  for (int i = tid; i < 8000; i += 256) a += p[i]*qw10[i/400];
  s[tid] = a; __syncthreads();
  for (int st = 128; st > 0; st >>= 1){
    if (tid < st) s[tid] += s[tid+st];
    __syncthreads();
  }
  if (tid == 0){
    float sc = 2.f*PI_F/20.f; sc *= sc;
    feat[bf] = s[0]*sc;
  }
}

__global__ void k_linear(const float* __restrict__ feat, const float* __restrict__ W,
                         const float* __restrict__ bvec, float* __restrict__ out){
  int t = blockIdx.x*blockDim.x + threadIdx.x;
  if (t >= 320) return;
  int b = t / 40, c = t % 40;
  float a = bvec[c];
  for (int f = 0; f < 100; ++f) a += feat[b*100+f]*W[f*40+c];
  out[t] = a;
}

// ---------------------------------------------------------------------------
// Host
// ---------------------------------------------------------------------------
static inline unsigned int cdivu(long long a, long long b){ return (unsigned int)((a + b - 1) / b); }

extern "C" void kernel_launch(void* const* d_in, const int* in_sizes, int n_in,
                              void* d_out, int out_size, void* d_ws, size_t ws_size,
                              hipStream_t stream){
  (void)in_sizes; (void)n_in; (void)out_size; (void)ws_size;
  const float* x        = (const float*)d_in[0];
  const float* w_s2     = (const float*)d_in[1];
  const float* bias_s2  = (const float*)d_in[2];
  const float* bn1_g    = (const float*)d_in[3];
  const float* bn1_b    = (const float*)d_in[4];
  const float* w_so3    = (const float*)d_in[5];
  const float* bias_so3 = (const float*)d_in[6];
  const float* bn2_g    = (const float*)d_in[7];
  const float* bn2_b    = (const float*)d_in[8];
  const float* lin_w    = (const float*)d_in[9];
  const float* lin_b    = (const float*)d_in[10];
  float* ws = (float*)d_ws;

  const size_t apackN = apack_off(10);      // halves per A plane (~1.16M)
  const size_t bpackN = bpack_off(10);      // halves per B plane (~13.5M)

  size_t p = 0;
  auto alloc = [&](size_t nf)->size_t { size_t o = p; p += (nf + 15) & ~(size_t)15; return o; };
  const size_t qw30  = alloc(60);
  const size_t qw16  = alloc(32);
  const size_t qw10  = alloc(20);
  const size_t d0s2  = alloc(60*256);        // d^l_{m,0}(beta60)
  const size_t dhalf = alloc(5456);          // d^l(pi/2), l<16
  const size_t d32   = alloc(32*5456);       // d^l(beta32), l<16
  const size_t d20   = alloc(20*1330);       // d^l(beta20), l<10
  const size_t img   = alloc(8*60*60);
  const size_t gs2   = alloc(2*8*60*60);     // complex
  const size_t xcf   = alloc(2*8*256);
  const size_t ycf   = alloc(2*100*256);
  const size_t wfc   = alloc(2*100*100*19);
  const size_t zc    = alloc(2*800*5456);    // complex; region reused for Z2
  const size_t mc    = alloc(2*800*32768);   // complex; M1 / G2 / M2
  const size_t hb    = alloc(800*32768);     // h / h2
  const size_t aprO  = alloc((apackN+1)/2);  // f16 fragment-packed GEMM operands
  const size_t apiO  = alloc((apackN+1)/2);
  const size_t bprO  = alloc((bpackN+1)/2);
  const size_t bpiO  = alloc((bpackN+1)/2);
  const size_t muO   = alloc(112);
  const size_t vaO   = alloc(112);
  const size_t feat  = alloc(800);
  const size_t z2O   = zc;                   // float2, 800*1330 complex (Zc consumed by then)

  // --- tables ---
  hipLaunchKernelGGL(k_qw, dim3(1), dim3(64), 0, stream, ws+qw30, 30);
  hipLaunchKernelGGL(k_qw, dim3(1), dim3(64), 0, stream, ws+qw16, 16);
  hipLaunchKernelGGL(k_qw, dim3(1), dim3(32), 0, stream, ws+qw10, 10);
  hipLaunchKernelGGL(k_wig_d0,   dim3(cdivu(60*256,256)),   dim3(256), 0, stream, ws+d0s2);
  hipLaunchKernelGGL(k_wig_pack, dim3(cdivu(5456,256)),     dim3(256), 0, stream, ws+dhalf, 1, 5456, 16, 1);
  hipLaunchKernelGGL(k_wig_pack, dim3(cdivu(32*5456,256)),  dim3(256), 0, stream, ws+d32, 32, 5456, 16, 0);
  hipLaunchKernelGGL(k_wig_pack, dim3(cdivu(20*1330,256)),  dim3(256), 0, stream, ws+d20, 20, 1330, 10, 0);

  // --- S2 stage ---
  hipMemsetAsync(ws+img, 0, 8*60*60*sizeof(float), stream);
  hipLaunchKernelGGL(k_project,   dim3(cdivu(8*1024,256)),  dim3(256), 0, stream, x, ws+img);
  hipLaunchKernelGGL(k_dft_alpha, dim3(cdivu(8*60*60,256)), dim3(256), 0, stream, ws+img, (float2*)(ws+gs2));
  hipLaunchKernelGGL(k_s2fft,  dim3(cdivu(8*256,256)),   dim3(256), 0, stream,
                     (const float2*)(ws+gs2), ws+qw30, ws+d0s2, (float2*)(ws+xcf));
  hipLaunchKernelGGL(k_s2rft,  dim3(cdivu(100*256,256)), dim3(256), 0, stream,
                     w_s2, ws+dhalf, (float2*)(ws+ycf));
  hipLaunchKernelGGL(k_zouter, dim3(cdivu(800*5456,256)), dim3(256), 0, stream,
                     (const float2*)(ws+xcf), (const float2*)(ws+ycf), (float2*)(ws+zc));
  hipLaunchKernelGGL(k_maccum, dim3(cdivu(800LL*32768,256)), dim3(256), 0, stream,
                     (const float2*)(ws+zc), ws+d32, (float2*)(ws+mc), 32, 16, 5456, 800LL*32768);
  hipLaunchKernelGGL(HIP_KERNEL_NAME(k_itile<32>), dim3(800*32), dim3(256), 0, stream,
                     (const float2*)(ws+mc), bias_s2, ws+hb);

  // --- BN + ReLU #1 ---
  hipLaunchKernelGGL(k_bnstats, dim3(100), dim3(256), 0, stream, ws+hb, ws+muO, ws+vaO, 32768, 8);
  hipLaunchKernelGGL(k_bnrelu,  dim3(cdivu(800LL*32768,256)), dim3(256), 0, stream,
                     ws+hb, ws+muO, ws+vaO, bn1_g, bn1_b, 32768, 800LL*32768);

  // --- SO3 stage ---
  hipLaunchKernelGGL(HIP_KERNEL_NAME(k_ftile<32>), dim3(800*32), dim3(256), 0, stream,
                     ws+hb, (float2*)(ws+mc));                                 // G2 reuses Mc
  hipMemsetAsync(ws+aprO, 0, apackN*sizeof(_Float16), stream);
  hipMemsetAsync(ws+apiO, 0, apackN*sizeof(_Float16), stream);
  hipMemsetAsync(ws+bprO, 0, bpackN*sizeof(_Float16), stream);
  hipMemsetAsync(ws+bpiO, 0, bpackN*sizeof(_Float16), stream);
  hipLaunchKernelGGL(k_x2a, dim3(cdivu(800*1330,256)), dim3(256), 0, stream,
                     (const float2*)(ws+mc), ws+qw16, ws+d32,
                     (_Float16*)(ws+aprO), (_Float16*)(ws+apiO));
  hipLaunchKernelGGL(k_wf,  dim3(cdivu(100*100*19,256)), dim3(256), 0, stream,
                     w_so3, (float2*)(ws+wfc));
  hipLaunchKernelGGL(k_y2b, dim3(cdivu(10000*1330,256)), dim3(256), 0, stream,
                     ws+dhalf, (const float2*)(ws+wfc),
                     (_Float16*)(ws+bprO), (_Float16*)(ws+bpiO));
  for (int l = 0; l < 10; ++l){
    int w = 2*l+1;
    int R = 8*w, C = 100*w;
    int KS = kp32(w) >> 5;
    hipLaunchKernelGGL(k_zgemm, dim3(cdivu(C,16), cdivu(R,16)), dim3(32), 0, stream,
                       (const _Float16*)(ws+aprO) + apack_off(l),
                       (const _Float16*)(ws+apiO) + apack_off(l),
                       (const _Float16*)(ws+bprO) + bpack_off(l),
                       (const _Float16*)(ws+bpiO) + bpack_off(l),
                       (float2*)(ws+z2O), l, R, C, KS);
  }
  hipLaunchKernelGGL(k_maccum, dim3(cdivu(800LL*8000,256)), dim3(256), 0, stream,
                     (const float2*)(ws+z2O), ws+d20, (float2*)(ws+mc), 20, 10, 1330, 800LL*8000);
  hipLaunchKernelGGL(HIP_KERNEL_NAME(k_itile<20>), dim3(800*20), dim3(128), 0, stream,
                     (const float2*)(ws+mc), bias_so3, ws+hb);                 // h2 reuses h

  // --- BN + ReLU #2, integrate, linear ---
  hipLaunchKernelGGL(k_bnstats, dim3(100), dim3(256), 0, stream, ws+hb, ws+muO, ws+vaO, 8000, 8);
  hipLaunchKernelGGL(k_bnrelu,  dim3(cdivu(800LL*8000,256)), dim3(256), 0, stream,
                     ws+hb, ws+muO, ws+vaO, bn2_g, bn2_b, 8000, 800LL*8000);
  hipLaunchKernelGGL(k_integrate, dim3(800), dim3(256), 0, stream, ws+hb, ws+qw10, ws+feat);
  hipLaunchKernelGGL(k_linear, dim3(2), dim3(256), 0, stream, ws+feat, lin_w, lin_b, (float*)d_out);
}
